// Net_88098369176511
// MI455X (gfx1250) — matrix operand order repaired and checked by reading, not compiled, not run on hardware
//
#include <hip/hip_runtime.h>
#include <hip/hip_bf16.h>

// ---------------------------------------------------------------------------
// CDNA5 (gfx1250) implementation. Strategy:
//  * All 32-in-channel convs / 1x1 GEMMs / the 4096x4096 cosine-similarity
//    Gram matrix run on V_WMMA_F32_16X16X32_F16 (wave32, M=16 outch tile,
//    N=16 pixel tile, K=32 in-channels == exactly one WMMA per tile).
//  * Activations kept in f16 NHWC so B-fragments (lane = pixel, 16 contiguous
//    channels per lane per half-wave) are single 32B loads; weights converted
//    once to f16 [tap][32][Cin] read as contiguous-K A-fragments
//    (lanes 0-15: K0-15, lanes 16-31: K16-31, same order as B).
//  * sim_block fuses WMMA Gram tiles with per-lane top-7 (D-layout: each lane
//    holds 8 q values for one fixed p; lanes l and l+16 cover complementary
//    q halves, merged via __shfl_xor(...,16)). Avoids the 256MB score matrix.
//  * Tiny FC / gather / mean / multi_block epilogues are scalar VALU kernels
//    (negligible FLOPs).
// ---------------------------------------------------------------------------

typedef __attribute__((ext_vector_type(16))) _Float16 v16h;
typedef __attribute__((ext_vector_type(8)))  _Float16 v8h;
typedef __attribute__((ext_vector_type(8)))  float    v8f;

#define HW   4096
#define TPB  256

__device__ __forceinline__ v8f wmma16(v16h a, v16h b, v8f c) {
  return __builtin_amdgcn_wmma_f32_16x16x32_f16(false, a, false, b, (short)0, c,
                                                false, false);
}

// A-fragment: A[m][k] = base[(row_base+m)*row_stride + kbase + k]
// lanes 0-15: m=lane, K=0..15; lanes 16-31: same m, K=16..31
__device__ __forceinline__ v16h load_afrag(const _Float16* base, int row_stride,
                                           int row_base, int kbase, int lane) {
  int r   = row_base + (lane & 15);
  int off = ((lane >> 4) << 4);
  const _Float16* p = base + (size_t)r * row_stride + kbase + off;
  v8h lo = *(const v8h*)(p);
  v8h hi = *(const v8h*)(p + 8);
  return __builtin_shufflevector(lo, hi, 0,1,2,3,4,5,6,7,8,9,10,11,12,13,14,15);
}

// B-fragment: B[k][n] = base[(col_base+n)*row_stride + kbase + k]
// lanes 0-15: n=lane, K=0..15; lanes 16-31: n=lane-16, K=16..31
__device__ __forceinline__ v16h load_bfrag(const _Float16* base, int row_stride,
                                           int col_base, int kbase, int lane) {
  int n  = col_base + (lane & 15);
  int kh = ((lane >> 4) << 4);
  return *(const v16h*)(base + (size_t)n * row_stride + kbase + kh);
}

// ------------------------- weight prep (f32 -> f16) -------------------------
// 3x3 conv weights  (Cout,Cin,3,3) f32  ->  [9][32][Cin] f16, rows>=Cout zero
__global__ void k_prep_w3(const float* __restrict__ w, _Float16* __restrict__ wh,
                          int Cout, int Cin, int total) {
  int t = blockIdx.x * blockDim.x + threadIdx.x;
  if (t >= total) return;
  int cc  = 32 * Cin;
  int tap = t / cc;
  int r   = t - tap * cc;
  int o   = r / Cin;
  int i   = r - o * Cin;
  float v = (o < Cout) ? w[(size_t)(o * Cin + i) * 9 + tap] : 0.f;
  wh[t] = (_Float16)v;
}

// 1x1 conv weights (Cout,Cin,1,1) f32 -> [32][Cin] f16
__global__ void k_prep_w1(const float* __restrict__ w, _Float16* __restrict__ wh,
                          int Cout, int Cin, int total) {
  int t = blockIdx.x * blockDim.x + threadIdx.x;
  if (t >= total) return;
  int o = t / Cin, i = t - o * Cin;
  wh[t] = (_Float16)((o < Cout) ? w[(size_t)o * Cin + i] : 0.f);
}

// f32 NCHW -> f16 NHWC (channel offset into wider buffer)
__global__ void k_nchw2nhwc(const float* __restrict__ in, _Float16* __restrict__ out,
                            int C, int P, int cstride, int coff, int total) {
  int t = blockIdx.x * blockDim.x + threadIdx.x;
  if (t >= total) return;
  int b = t / (C * P);
  int r = t - b * C * P;
  int c = r / P;
  int p = r - c * P;
  out[((size_t)b * P + p) * cstride + coff + c] = (_Float16)in[t];
}

// ------------------- direct conv 3x3 for tiny Cin (3 or 6) ------------------
__global__ void k_conv_small(const float* __restrict__ in, int bufC, int Cin,
                             const float* __restrict__ w, const float* __restrict__ bias,
                             _Float16* __restrict__ out, int ostride, int total) {
  int t = blockIdx.x * blockDim.x + threadIdx.x;
  if (t >= total) return;
  int b = t >> 12;
  int p = t & 4095;
  int y = p >> 6, x = p & 63;
  float acc[32];
#pragma unroll
  for (int o = 0; o < 32; ++o) acc[o] = bias[o];
  const float* inb = in + (size_t)b * bufC * HW;
  for (int ky = 0; ky < 3; ++ky) {
    int yy = y + ky - 1;
    if ((unsigned)yy >= 64u) continue;
    for (int kx = 0; kx < 3; ++kx) {
      int xx = x + kx - 1;
      if ((unsigned)xx >= 64u) continue;
      const float* ip = inb + (yy << 6) + xx;
      for (int i = 0; i < Cin; ++i) {
        float v = ip[(size_t)i * HW];
        const float* wr = w + (size_t)(i * 3 + ky) * 3 + kx;
#pragma unroll
        for (int o = 0; o < 32; ++o) acc[o] += v * wr[(size_t)o * Cin * 9];
      }
    }
  }
  _Float16* orow = out + (size_t)t * ostride;
#pragma unroll
  for (int o = 0; o < 32; ++o) orow[o] = (_Float16)fmaxf(acc[o], 0.f);
}

// ------------------ WMMA 3x3 conv, Cin=32, Cout<=32, any stride/dil ---------
__global__ void k_conv3(const _Float16* __restrict__ in, int instride,
                        const _Float16* __restrict__ wh, const float* __restrict__ bias,
                        _Float16* __restrict__ out, int ostride, int ooff, int Cout,
                        int Hi, int Wi, int Ho, int lwo, int stride, int dil, int pad,
                        int relu_, int nwaves) {
  int wave = (blockIdx.x * blockDim.x + threadIdx.x) >> 5;
  if (wave >= nwaves) return;
  int lane   = threadIdx.x & 31;
  int Wo     = 1 << lwo;
  int ptiles = (Ho << lwo) >> 4;
  int b      = wave / ptiles;
  int pbase  = (wave - b * ptiles) << 4;
  int kh     = ((lane >> 4) << 4);
  int p      = pbase + (lane & 15);
  int oy = p >> lwo, ox = p & (Wo - 1);
  int iy0 = oy * stride - pad, ix0 = ox * stride - pad;
  const _Float16* inb = in + (size_t)b * (Hi * Wi) * instride;
  v8f acc0 = {};
  v8f acc1 = {};
#pragma unroll
  for (int ky = 0; ky < 3; ++ky) {
#pragma unroll
    for (int kx = 0; kx < 3; ++kx) {
      int tap = ky * 3 + kx;
      int iy = iy0 + ky * dil, ix = ix0 + kx * dil;
      v16h bf = {};
      if (((unsigned)iy < (unsigned)Hi) & ((unsigned)ix < (unsigned)Wi))
        bf = *(const v16h*)(inb + (size_t)(iy * Wi + ix) * instride + kh);
      const _Float16* wt = wh + tap * 32 * 32;
      v16h a0 = load_afrag(wt, 32, 0, 0, lane);
      v16h a1 = load_afrag(wt, 32, 16, 0, lane);
      acc0 = wmma16(a0, bf, acc0);
      acc1 = wmma16(a1, bf, acc1);
    }
  }
  int mrow = ((lane >> 4) << 3);
  _Float16* orow = out + ((size_t)b * (Ho << lwo) + p) * ostride + ooff;
#pragma unroll
  for (int v = 0; v < 8; ++v) {
    int o0 = mrow + v, o1 = 16 + mrow + v;
    float f0 = acc0[v] + bias[o0];
    float f1 = acc1[v] + bias[o1];
    if (relu_) { f0 = fmaxf(f0, 0.f); f1 = fmaxf(f1, 0.f); }
    if (o0 < Cout) orow[o0] = (_Float16)f0;
    if (o1 < Cout) orow[o1] = (_Float16)f1;
  }
}

// -------- WMMA 1x1 GEMM over 4096 pixels: Cout=32, Cin in {32,64,96} --------
__global__ void k_gemm1(const _Float16* __restrict__ in, int instride, int Cin,
                        const _Float16* __restrict__ wh, const float* __restrict__ bias,
                        _Float16* __restrict__ out,
                        const _Float16* __restrict__ resid, const float* __restrict__ wscalar,
                        int relu_, int nwaves) {
  int wave = (blockIdx.x * blockDim.x + threadIdx.x) >> 5;
  if (wave >= nwaves) return;
  int lane  = threadIdx.x & 31;
  int b     = wave >> 8;
  int pbase = (wave & 255) << 4;
  int kh    = ((lane >> 4) << 4);
  int p     = pbase + (lane & 15);
  const _Float16* inr = in + ((size_t)b * HW + p) * instride;
  v8f acc0 = {};
  v8f acc1 = {};
  for (int kt = 0; kt < Cin; kt += 32) {
    v16h bf = *(const v16h*)(inr + kt + kh);
    v16h a0 = load_afrag(wh, Cin, 0, kt, lane);
    v16h a1 = load_afrag(wh, Cin, 16, kt, lane);
    acc0 = wmma16(a0, bf, acc0);
    acc1 = wmma16(a1, bf, acc1);
  }
  float wres = resid ? wscalar[0] : 0.f;
  int mrow = ((lane >> 4) << 3);
  _Float16* orow = out + ((size_t)b * HW + p) * 32;
  const _Float16* rrow = resid ? resid + ((size_t)b * HW + p) * 32 : (const _Float16*)0;
#pragma unroll
  for (int v = 0; v < 8; ++v) {
    int o0 = mrow + v, o1 = 16 + mrow + v;
    float f0 = acc0[v] + bias[o0];
    float f1 = acc1[v] + bias[o1];
    if (rrow) { f0 += wres * (float)rrow[o0]; f1 += wres * (float)rrow[o1]; }
    if (relu_) { f0 = fmaxf(f0, 0.f); f1 = fmaxf(f1, 0.f); }
    orow[o0] = (_Float16)f0;
    orow[o1] = (_Float16)f1;
  }
}

// ----------------------- cosine normalize (per pixel) -----------------------
__global__ void k_xnorm(const _Float16* __restrict__ in, _Float16* __restrict__ out,
                        int total) {
  int t = blockIdx.x * blockDim.x + threadIdx.x;
  if (t >= total) return;
  const _Float16* r = in + (size_t)t * 32;
  float s = 0.f;
#pragma unroll
  for (int i = 0; i < 32; ++i) { float v = (float)r[i]; s += v * v; }
  float inv = rsqrtf(s);
  _Float16* o = out + (size_t)t * 32;
#pragma unroll
  for (int i = 0; i < 32; ++i) o[i] = (_Float16)((float)r[i] * inv);
}

// ---- fused WMMA Gram matrix + per-lane top-7 (no 4096x4096 materialize) ----
__global__ void k_simtopk(const _Float16* __restrict__ Xn, float* __restrict__ tvout,
                          int* __restrict__ idxout, int nwaves) {
  int wave = (blockIdx.x * blockDim.x + threadIdx.x) >> 5;
  if (wave >= nwaves) return;
  int lane  = threadIdx.x & 31;
  int b     = wave >> 8;
  int pbase = (wave & 255) << 4;
  const _Float16* Xb = Xn + (size_t)b * HW * 32;
  v16h bf = load_bfrag(Xb, 32, pbase, 0, lane);  // fixed p columns
  float tv[7];
  int   ti[7];
#pragma unroll
  for (int k = 0; k < 7; ++k) { tv[k] = -3.0e38f; ti[k] = 0; }
  for (int qt = 0; qt < 256; ++qt) {
    int qbase = qt << 4;
    v16h af = load_afrag(Xb, 32, qbase, 0, lane);  // q rows
    v8f z = {};
    v8f d = wmma16(af, bf, z);                     // d[v] = score[q][p]
    int q0 = qbase + ((lane >> 4) << 3);
#pragma unroll
    for (int v = 0; v < 8; ++v) {
      float val = d[v];
      if (val > tv[6]) {
        tv[6] = val; ti[6] = q0 + v;
#pragma unroll
        for (int k = 6; k > 0; --k) {
          if (tv[k] > tv[k - 1]) {
            float tf = tv[k]; tv[k] = tv[k - 1]; tv[k - 1] = tf;
            int   tj = ti[k]; ti[k] = ti[k - 1]; ti[k - 1] = tj;
          }
        }
      }
    }
  }
  // merge complementary q-halves: lane l <-> lane l+16 (snapshot first!)
  float ptv[7];
  int   pti[7];
#pragma unroll
  for (int k = 0; k < 7; ++k) {
    ptv[k] = __shfl_xor(tv[k], 16);
    pti[k] = __shfl_xor(ti[k], 16);
  }
#pragma unroll
  for (int k = 0; k < 7; ++k) {
    float val = ptv[k];
    if (val > tv[6]) {
      tv[6] = val; ti[6] = pti[k];
#pragma unroll
      for (int j = 6; j > 0; --j) {
        if (tv[j] > tv[j - 1]) {
          float tf = tv[j]; tv[j] = tv[j - 1]; tv[j - 1] = tf;
          int   tj = ti[j]; ti[j] = ti[j - 1]; ti[j - 1] = tj;
        }
      }
    }
  }
  if (lane < 16) {
    int p = pbase + lane;
    size_t base = ((size_t)b * HW + p) * 7;
#pragma unroll
    for (int k = 0; k < 7; ++k) { tvout[base + k] = tv[k]; idxout[base + k] = ti[k]; }
  }
}

// top-7 gather + mix einsum; writes f16 NHWC and f32 NCHW (fr/fs output)
__global__ void k_simmix(const _Float16* __restrict__ X, const float* __restrict__ tv,
                         const int* __restrict__ idx, const float* __restrict__ wmix,
                         const float* __restrict__ bmix, _Float16* __restrict__ out16,
                         float* __restrict__ out32, int total) {
  int t = blockIdx.x * blockDim.x + threadIdx.x;
  if (t >= total) return;
  int b = t >> 12;
  int p = t & 4095;
  float acc[32];
#pragma unroll
  for (int o = 0; o < 32; ++o) acc[o] = bmix[o];
  size_t tb = ((size_t)b * HW + p) * 7;
  for (int k = 0; k < 7; ++k) {
    float tvk = tv[tb + k];
    int   q   = idx[tb + k];
    const _Float16* col = X + ((size_t)b * HW + q) * 32;
    for (int i = 0; i < 32; ++i) {
      float c = tvk * (float)col[i];
      const float* wr = wmix + (size_t)i * 7 + k;  // w[o][i][k], o-stride 224
#pragma unroll
      for (int o = 0; o < 32; ++o) acc[o] += c * wr[(size_t)o * 224];
    }
  }
  _Float16* orow = out16 + (size_t)t * 32;
  float* ob = out32 + (size_t)b * 32 * HW + p;
#pragma unroll
  for (int o = 0; o < 32; ++o) {
    float f = fmaxf(acc[o], 0.f);
    orow[o] = (_Float16)f;
    ob[(size_t)o * HW] = f;
  }
}

// mean over 4x4 spatial of final estimator map -> (B,32) f32
__global__ void k_mean(const _Float16* __restrict__ in, float* __restrict__ e, int total) {
  int t = blockIdx.x * blockDim.x + threadIdx.x;
  if (t >= total) return;
  int b = t >> 5, c = t & 31;
  const _Float16* r = in + (size_t)b * 16 * 32 + c;
  float s = 0.f;
#pragma unroll
  for (int p = 0; p < 16; ++p) s += (float)r[p * 32];
  e[t] = s * (1.f / 16.f);
}

__global__ void k_concat2(const float* __restrict__ a, const float* __restrict__ c,
                          float* __restrict__ out, int total) {
  int t = blockIdx.x * blockDim.x + threadIdx.x;
  if (t >= total) return;
  int b = t >> 6, j = t & 63;
  out[t] = (j < 32) ? a[b * 32 + j] : c[b * 32 + j - 32];
}

// tiny dense FC: y[b][o] = act(w[o]·x[b] + bias[o]); optional mirrored copy
__global__ void k_fc(const float* __restrict__ x, const float* __restrict__ w,
                     const float* __restrict__ bias, float* __restrict__ y,
                     float* __restrict__ y2, int Cin, int Cout, int relu_, int total) {
  int t = blockIdx.x * blockDim.x + threadIdx.x;
  if (t >= total) return;
  int b = t / Cout, o = t - b * Cout;
  const float* wr = w + (size_t)o * Cin;
  const float* xr = x + (size_t)b * Cin;
  float s = bias[o];
  for (int i = 0; i < Cin; ++i) s += wr[i] * xr[i];
  if (relu_) s = fmaxf(s, 0.f);
  y[t] = s;
  if (y2) y2[t] = s;
}

// multi_block: out[b,o,p] = sum_k M[k,o,p] * (sum_i patch[k,i,p]*K[k,i,o]) + x[:, :3]
__global__ void k_multi(const _Float16* __restrict__ feat, const _Float16* __restrict__ M,
                        const float* __restrict__ K, const float* __restrict__ xres,
                        float* __restrict__ out, int total) {
  int t = blockIdx.x * blockDim.x + threadIdx.x;
  if (t >= total) return;
  int b = t >> 12;
  int p = t & 4095;
  int y = p >> 6, x = p & 63;
  const float* Kbase = K + (size_t)b * 864;
  const _Float16* Mrow = M + (size_t)t * 32;
  float a0 = 0.f, a1 = 0.f, a2 = 0.f;
#pragma unroll
  for (int ky = 0; ky < 3; ++ky) {
#pragma unroll
    for (int kx = 0; kx < 3; ++kx) {
      int kk = ky * 3 + kx;
      float m0 = (float)Mrow[kk * 3 + 0];
      float m1 = (float)Mrow[kk * 3 + 1];
      float m2 = (float)Mrow[kk * 3 + 2];
      int yy = y + ky - 1, xx = x + kx - 1;
      float s0 = 0.f, s1 = 0.f, s2 = 0.f;
      if (((unsigned)yy < 64u) & ((unsigned)xx < 64u)) {
        const _Float16* col = feat + (((size_t)b << 12) + (yy << 6) + xx) * 32;
        const float* Kp = Kbase + kk * 96;
        for (int i = 0; i < 32; ++i) {
          float pv = (float)col[i];
          s0 += pv * Kp[i * 3 + 0];
          s1 += pv * Kp[i * 3 + 1];
          s2 += pv * Kp[i * 3 + 2];
        }
      }
      a0 += s0 * m0; a1 += s1 * m1; a2 += s2 * m2;
    }
  }
  const float* xr = xres + (size_t)b * 6 * HW;
  out[((size_t)b * 3 + 0) * HW + p] = a0 + xr[0 * HW + p];
  out[((size_t)b * 3 + 1) * HW + p] = a1 + xr[1 * HW + p];
  out[((size_t)b * 3 + 2) * HW + p] = a2 + xr[2 * HW + p];
}

// ---------------------------------------------------------------------------
extern "C" void kernel_launch(void* const* d_in, const int* in_sizes, int n_in,
                              void* d_out, int out_size, void* d_ws, size_t ws_size,
                              hipStream_t stream) {
  (void)out_size;
  if (n_in < 98) return;
  const float* real   = (const float*)d_in[0];
  const float* syn    = (const float*)d_in[1];
  const float* f_real = (const float*)d_in[2];
  const float* f_syn  = (const float*)d_in[3];
  const int B = in_sizes[0] / (6 * HW);
  float* dout = (float*)d_out;

  auto P = [&](int i) { return (const float*)d_in[i]; };
  // param leaf order = jax pytree (sorted dict keys), after the 4 data inputs:
  const float* Rec_b = P(4); const float* Rec_w = P(5);
  struct ResP { const float *W, *b1, *b2, *b3, *bm, *w1, *w2, *w3, *wm; };
  auto getres = [&](int base) {
    ResP r{P(base), P(base + 1), P(base + 2), P(base + 3), P(base + 4),
           P(base + 5), P(base + 6), P(base + 7), P(base + 8)};
    return r;
  };
  ResP gf0 = getres(6), gf1 = getres(15);
  const float* gfil0_b = P(24); const float* gfil0_w = P(25);
  const float* gfil1_b = P(26); const float* gfil1_w = P(27);
  const float* inp_b = P(28);   const float* inp_w = P(29);
  ResP mn0 = getres(30), mn1 = getres(39), mn2 = getres(48), mn3 = getres(57);
  const float* mix_b = P(66);   const float* mix_w = P(67);
  const float* est_b[8]; const float* est_w[8];
  for (int i = 0; i < 8; ++i) { est_b[i] = P(68 + 2 * i); est_w[i] = P(69 + 2 * i); }
  const float *ffc1_b = P(84), *ffc1_w = P(85), *ffc2_b = P(86), *ffc2_w = P(87);
  const float *fc1_b = P(88), *fc1_w = P(89), *fc2_b = P(90), *fc2_w = P(91);
  const float *nein_b = P(92), *nein_w = P(93);
  const float *Kb = P(94), *Kw = P(95), *Mb = P(96), *Mw = P(97);

  // ----------------------------- workspace ---------------------------------
  char* wsp = (char*)d_ws;
  size_t off = 0;
  auto alloc = [&](size_t bytes) -> void* {
    void* r = wsp + off;
    off = (off + bytes + 255) & ~(size_t)255;
    return r;
  };
  const size_t F16MAP = (size_t)B * HW * 32 * sizeof(_Float16);
  _Float16* wh_est[8];
  for (int i = 0; i < 8; ++i) wh_est[i] = (_Float16*)alloc(9 * 32 * 32 * 2);
  _Float16* wh_res[6][3];
  _Float16* wh1_res[6];
  for (int r = 0; r < 6; ++r) {
    for (int j = 0; j < 3; ++j) wh_res[r][j] = (_Float16*)alloc(9 * 32 * 32 * 2);
    wh1_res[r] = (_Float16*)alloc(32 * 96 * 2);
  }
  _Float16* wh_fil0 = (_Float16*)alloc(9 * 32 * 32 * 2);
  _Float16* wh_fil1 = (_Float16*)alloc(9 * 32 * 32 * 2);
  _Float16* wh_Mw   = (_Float16*)alloc(9 * 32 * 32 * 2);
  _Float16* wh_Rec  = (_Float16*)alloc(32 * 64 * 2);
  _Float16* areal = (_Float16*)alloc(F16MAP);
  _Float16* asyn  = (_Float16*)alloc(F16MAP);
  _Float16* estA  = (_Float16*)alloc(F16MAP);
  _Float16* estB  = (_Float16*)alloc(F16MAP);
  float* e_real = (float*)alloc(B * 32 * 4);
  float* e_syn  = (float*)alloc(B * 32 * 4);
  float* ecat   = (float*)alloc(B * 64 * 4);
  float* hbuf   = (float*)alloc(B * 64 * 4);
  float* t1buf  = (float*)alloc(B * 32 * 4);
  float* CFr    = (float*)alloc(B * 32 * 4);
  float* CFs    = (float*)alloc(B * 32 * 4);
  _Float16* in64  = (_Float16*)alloc((size_t)B * HW * 64 * 2);
  _Float16* featA = (_Float16*)alloc(F16MAP);
  _Float16* featB = (_Float16*)alloc(F16MAP);
  _Float16* ybuf  = (_Float16*)alloc((size_t)B * HW * 96 * 2);
  _Float16* Xn    = (_Float16*)alloc(F16MAP);
  float* tvb  = (float*)alloc((size_t)B * HW * 7 * 4);
  int*   idxb = (int*)alloc((size_t)B * HW * 7 * 4);
  _Float16* xsim  = (_Float16*)alloc(F16MAP);
  _Float16* filtA = (_Float16*)alloc(F16MAP);
  _Float16* filtB = (_Float16*)alloc(F16MAP);
  _Float16* Mbuf  = (_Float16*)alloc(F16MAP);
  float* Kbuf = (float*)alloc((size_t)B * 864 * 4);
  if (off > ws_size) return;

  auto g1 = [&](int total) { return dim3((unsigned)((total + TPB - 1) / TPB)); };
  auto gw = [&](int nwaves) { return dim3((unsigned)((nwaves + 7) / 8)); };

  // ------------------------- weight prep launches --------------------------
  {
    ResP resv[6] = {mn0, mn1, mn2, mn3, gf0, gf1};
    int tw3 = 9 * 32 * 32;
    for (int r = 0; r < 6; ++r) {
      const float* w3s[3] = {resv[r].w1, resv[r].w2, resv[r].w3};
      for (int j = 0; j < 3; ++j)
        k_prep_w3<<<g1(tw3), TPB, 0, stream>>>(w3s[j], wh_res[r][j], 32, 32, tw3);
      k_prep_w1<<<g1(32 * 96), TPB, 0, stream>>>(resv[r].wm, wh1_res[r], 32, 96, 32 * 96);
    }
    for (int i = 0; i < 8; ++i)
      k_prep_w3<<<g1(tw3), TPB, 0, stream>>>(est_w[i], wh_est[i], 32, 32, tw3);
    k_prep_w3<<<g1(tw3), TPB, 0, stream>>>(gfil0_w, wh_fil0, 32, 32, tw3);
    k_prep_w3<<<g1(tw3), TPB, 0, stream>>>(gfil1_w, wh_fil1, 32, 32, tw3);
    k_prep_w3<<<g1(tw3), TPB, 0, stream>>>(Mw, wh_Mw, 27, 32, tw3);
    k_prep_w1<<<g1(32 * 64), TPB, 0, stream>>>(Rec_w, wh_Rec, 32, 64, 32 * 64);
  }

  // --------------------------------- ne ------------------------------------
  k_conv_small<<<g1(B * HW), TPB, 0, stream>>>(real, 6, 3, nein_w, nein_b, areal, 32, B * HW);
  k_conv_small<<<g1(B * HW), TPB, 0, stream>>>(syn, 6, 3, nein_w, nein_b, asyn, 32, B * HW);

  auto run_est = [&](_Float16* src, float* eout) {
    int Hi = 64, Wi = 64;
    _Float16* cur = src;
    _Float16* bufs[2] = {estA, estB};
    for (int L = 0; L < 8; ++L) {
      int stride = (L & 1) ? 2 : 1;
      int Ho = Hi / stride, Wo = Wi / stride;
      int lwo = 0;
      while ((1 << lwo) < Wo) ++lwo;
      _Float16* dst = bufs[L & 1];
      int nwaves = B * ((Ho * Wo) >> 4);
      k_conv3<<<gw(nwaves), TPB, 0, stream>>>(cur, 32, wh_est[L], est_b[L], dst, 32, 0, 32,
                                              Hi, Wi, Ho, lwo, stride, 1, 1, 1, nwaves);
      cur = dst; Hi = Ho; Wi = Wo;
    }
    k_mean<<<g1(B * 32), TPB, 0, stream>>>(cur, eout, B * 32);
  };
  run_est(areal, e_real);
  run_est(asyn, e_syn);

  // output offsets in d_out (concatenated tuple, flat f32)
  const size_t o_sout = (size_t)B * 3 * HW;
  const size_t o_fr   = 2 * o_sout;
  const size_t o_fs   = o_fr + (size_t)B * 32 * HW;
  const size_t o_n    = o_fs + (size_t)B * 32 * HW;
  const size_t o_cfr  = o_n + (size_t)B * 2;
  const size_t o_cfs  = o_cfr + (size_t)B * 32;

  k_concat2<<<g1(B * 64), TPB, 0, stream>>>(e_real, e_syn, ecat, B * 64);
  k_fc<<<g1(B * 64), TPB, 0, stream>>>(ecat, fc1_w, fc1_b, hbuf, (float*)0, 64, 64, 1, B * 64);
  k_fc<<<g1(B * 2), TPB, 0, stream>>>(hbuf, fc2_w, fc2_b, dout + o_n, (float*)0, 64, 2, 0, B * 2);
  k_fc<<<g1(B * 32), TPB, 0, stream>>>(e_real, ffc1_w, ffc1_b, t1buf, (float*)0, 32, 32, 1, B * 32);
  k_fc<<<g1(B * 32), TPB, 0, stream>>>(t1buf, ffc2_w, ffc2_b, CFr, dout + o_cfr, 32, 32, 1, B * 32);
  k_fc<<<g1(B * 32), TPB, 0, stream>>>(e_syn, ffc1_w, ffc1_b, t1buf, (float*)0, 32, 32, 1, B * 32);
  k_fc<<<g1(B * 32), TPB, 0, stream>>>(t1buf, ffc2_w, ffc2_b, CFs, dout + o_cfs, 32, 32, 1, B * 32);

  // ------------------------------ reduction --------------------------------
  ResP resv[6] = {mn0, mn1, mn2, mn3, gf0, gf1};
  auto reduction = [&](const float* x6, const float* feat_in, const float* CF,
                       float* rout, float* fr) {
    const int NW = B * 256;  // waves over (B, 4096/16) tiles
    k_conv_small<<<g1(B * HW), TPB, 0, stream>>>(x6, 6, 6, inp_w, inp_b, in64, 64, B * HW);
    k_nchw2nhwc<<<g1(B * 32 * HW), TPB, 0, stream>>>(feat_in, in64, 32, HW, 64, 32, B * 32 * HW);
    k_gemm1<<<gw(NW), TPB, 0, stream>>>(in64, 64, 64, wh_Rec, Rec_b, featA,
                                        (const _Float16*)0, (const float*)0, 1, NW);
    _Float16* cur = featA;
    _Float16* nxt = featB;
    auto resblock = [&](int rb) {
      const ResP& rp = resv[rb];
      k_conv3<<<gw(NW), TPB, 0, stream>>>(cur, 32, wh_res[rb][0], rp.b1, ybuf, 96, 0, 32,
                                          64, 64, 64, 6, 1, 1, 1, 1, NW);
      k_conv3<<<gw(NW), TPB, 0, stream>>>(cur, 32, wh_res[rb][1], rp.b2, ybuf, 96, 32, 32,
                                          64, 64, 64, 6, 1, 2, 2, 1, NW);
      k_conv3<<<gw(NW), TPB, 0, stream>>>(cur, 32, wh_res[rb][2], rp.b3, ybuf, 96, 64, 32,
                                          64, 64, 64, 6, 1, 3, 3, 1, NW);
      k_gemm1<<<gw(NW), TPB, 0, stream>>>(ybuf, 96, 96, wh1_res[rb], rp.bm, nxt,
                                          cur, rp.W, 0, NW);
      _Float16* t = cur; cur = nxt; nxt = t;
    };
    resblock(0); resblock(1); resblock(2); resblock(3);
    // sim_block
    k_xnorm<<<g1(B * HW), TPB, 0, stream>>>(cur, Xn, B * HW);
    k_simtopk<<<gw(NW), TPB, 0, stream>>>(Xn, tvb, idxb, NW);
    k_simmix<<<g1(B * HW), TPB, 0, stream>>>(cur, tvb, idxb, mix_w, mix_b, xsim, fr, B * HW);
    // get_feature
    cur = xsim; nxt = featA;
    resblock(4); resblock(5);
    _Float16* featF = cur;
    // get_filters
    k_conv3<<<gw(NW), TPB, 0, stream>>>(featF, 32, wh_fil0, gfil0_b, filtA, 32, 0, 32,
                                        64, 64, 64, 6, 1, 1, 1, 1, NW);
    k_conv3<<<gw(NW), TPB, 0, stream>>>(filtA, 32, wh_fil1, gfil1_b, filtB, 32, 0, 32,
                                        64, 64, 64, 6, 1, 1, 1, 1, NW);
    // multi_block: M (27-ch conv, no relu), K (CF @ Kw), final einsum + residual
    k_conv3<<<gw(NW), TPB, 0, stream>>>(filtB, 32, wh_Mw, Mb, Mbuf, 32, 0, 27,
                                        64, 64, 64, 6, 1, 1, 1, 0, NW);
    k_fc<<<g1(B * 864), TPB, 0, stream>>>(CF, Kw, Kb, Kbuf, (float*)0, 32, 864, 0, B * 864);
    k_multi<<<g1(B * HW), TPB, 0, stream>>>(featF, Mbuf, Kbuf, x6, rout, B * HW);
  };
  reduction(real, f_real, CFr, dout + 0, dout + o_fr);
  reduction(syn, f_syn, CFs, dout + o_sout, dout + o_fs);
}
